// TransformerBlock_18700287606948
// MI455X (gfx1250) — matrix operand order repaired and checked by reading, not compiled, not run on hardware
//
#include <hip/hip_runtime.h>
#include <hip/hip_bf16.h>

typedef _Float16 half_t;
typedef __attribute__((ext_vector_type(16))) _Float16 v16h;
typedef __attribute__((ext_vector_type(8)))  _Float16 v8h;
typedef __attribute__((ext_vector_type(8)))  float    v8f;

#define DIM   1024
#define NHEAD 16
#define HDIM  64
#define FFN   4096
#define BB    2
#define TT    2048
#define MROWS (BB*TT)

// ---------------------------------------------------------------------------
// CDNA5 data movers.
// Low 32 bits of a generic pointer to __shared__ are the LDS byte address.
// ---------------------------------------------------------------------------
#define USE_ASYNC_LDS 1

__device__ __forceinline__ void async_copy_b128(void* lds_dst, const void* gsrc) {
    uint32_t ldsoff = (uint32_t)(uintptr_t)lds_dst;
    uint64_t ga = (uint64_t)(uintptr_t)gsrc;
    asm volatile("global_load_async_to_lds_b128 %0, %1, off"
                 :: "v"(ldsoff), "v"(ga) : "memory");
}
__device__ __forceinline__ void wait_async0() {
    asm volatile("s_wait_asynccnt 0x0" ::: "memory");
}

#if defined(__has_builtin)
#if __has_builtin(__builtin_amdgcn_tensor_load_to_lds)
#define HAVE_TDM 1
#endif
#endif

#ifdef HAVE_TDM
typedef __attribute__((ext_vector_type(4))) unsigned int tdm_u32x4;
typedef __attribute__((ext_vector_type(8))) int tdm_i32x8;
typedef __attribute__((ext_vector_type(4))) int tdm_i32x4;

// 2-D tile load: tile_d0 (contiguous, elems) x tile_d1 (rows), 2-byte elems,
// row stride = stride0 elems. LDS rows padded: every 16 DWORDs (+32 halves)
// insert 4 DWORDs -> 40-half LDS row pitch.
__device__ __forceinline__ void tdm_load_2d(void* lds_dst, const void* gsrc,
                                            int tile_d0, int tile_d1,
                                            int tensor_d0, int tensor_d1,
                                            long stride0) {
    const uint64_t ga = (uint64_t)(uintptr_t)gsrc;
    tdm_u32x4 g0;
    g0[0] = 1u;                               // count=1, user mode, no gather
    g0[1] = (uint32_t)(uintptr_t)lds_dst;     // lds_addr
    g0[2] = (uint32_t)ga;                     // global_addr[31:0]
    g0[3] = (uint32_t)((ga >> 32) & 0x1FFFFFFu) | (2u << 30);  // addr hi, type=2
    tdm_i32x8 g1;
    g1[0] = (int)((1u << 16)      // data_size = 2 bytes
                  | (1u << 20)    // pad_enable
                  | (3u << 22)    // pad_interval: 16 DWORDs
                  | (3u << 25));  // pad_amount: 4 DWORDs
    g1[1] = (int)((uint32_t)tensor_d0 << 16);
    g1[2] = (int)(((uint32_t)tensor_d0 >> 16) | ((uint32_t)tensor_d1 << 16));
    g1[3] = (int)((((uint32_t)tensor_d1 >> 16) & 0xFFFFu) |
                  ((uint32_t)tile_d0 << 16));
    g1[4] = (int)(uint32_t)tile_d1;           // tile_dim1 (tile_dim2 = 0)
    g1[5] = (int)(uint32_t)(stride0 & 0xFFFFFFFFl);
    g1[6] = (int)(uint32_t)((stride0 >> 32) & 0xFFFFl);
    g1[7] = 0;
    tdm_i32x4 z4 = {};
#if __has_include(<hip/amd_detail/amd_gfx1250_TDM.h>)
    tdm_i32x8 z8 = {};
    __builtin_amdgcn_tensor_load_to_lds(g0, g1, z4, z4, z8, 0);
#else
    __builtin_amdgcn_tensor_load_to_lds(g0, g1, z4, z4, 0);
#endif
}

__device__ __forceinline__ void wait_tensor0() {
#if __has_builtin(__builtin_amdgcn_s_wait_tensorcnt)
    __builtin_amdgcn_s_wait_tensorcnt(0);
#else
    asm volatile("s_wait_tensorcnt 0x0" ::: "memory");
#endif
}
#endif  // HAVE_TDM

__device__ __forceinline__ v16h join16(v8h lo, v8h hi) {
    return __builtin_shufflevector(lo, hi, 0, 1, 2, 3, 4, 5, 6, 7,
                                   8, 9, 10, 11, 12, 13, 14, 15);
}

// ---------------------------------------------------------------------------
// fused fp32 cast + transpose: W[K,N] fp32 -> Wt[N,K] f16   (grid: K/32, N/32)
// ---------------------------------------------------------------------------
__global__ __launch_bounds__(256) void cast_transpose_kernel(
    const float* __restrict__ W, half_t* __restrict__ Wt, int K, int N) {
    __shared__ float tile[32][33];
    const int k0 = blockIdx.x * 32, n0 = blockIdx.y * 32;
    const int tx = threadIdx.x & 31, ty = threadIdx.x >> 5;
#pragma unroll
    for (int i = 0; i < 4; ++i)
        tile[ty + 8 * i][tx] = W[(size_t)(k0 + ty + 8 * i) * N + n0 + tx];
    __syncthreads();
#pragma unroll
    for (int i = 0; i < 4; ++i)
        Wt[(size_t)(n0 + ty + 8 * i) * K + k0 + tx] = (half_t)tile[tx][ty + 8 * i];
}

// ---------------------------------------------------------------------------
// RMSNorm: row of 1024 fp32 -> f16
// ---------------------------------------------------------------------------
__global__ __launch_bounds__(256) void rmsnorm_kernel(const float* __restrict__ x,
                                                      const float* __restrict__ w,
                                                      half_t* __restrict__ out) {
    __shared__ float red[256];
    const int row = blockIdx.x;
    const float* xr = x + (size_t)row * DIM;
    float vals[4];
    float ss = 0.f;
#pragma unroll
    for (int i = 0; i < 4; ++i) {
        vals[i] = xr[threadIdx.x + 256 * i];
        ss += vals[i] * vals[i];
    }
    red[threadIdx.x] = ss;
    __syncthreads();
    for (int off = 128; off > 0; off >>= 1) {
        if (threadIdx.x < off) red[threadIdx.x] += red[threadIdx.x + off];
        __syncthreads();
    }
    const float inv = rsqrtf(red[0] * (1.0f / DIM) + 1e-5f);
#pragma unroll
    for (int i = 0; i < 4; ++i) {
        int c = threadIdx.x + 256 * i;
        out[(size_t)row * DIM + c] = (half_t)(vals[i] * inv * w[c]);
    }
}

// ---------------------------------------------------------------------------
// WMMA GEMM (TN): C[M,N] = A[M,K](f16) @ Bt[N,K](f16)^T + bias (+residual)
// 256 threads = 8 waves (2x4), block tile 64x128, wave tile 32x32, K-step 32.
// Double-buffered LDS fed by the Tensor Data Mover (fallback: async copies).
// ---------------------------------------------------------------------------
template <bool RES, bool OUT32, bool OUT16>
__global__ __launch_bounds__(256) void gemm_kernel(const half_t* __restrict__ A,
                                                   const half_t* __restrict__ Bt,
                                                   const float* __restrict__ bias,
                                                   const float* __restrict__ residual,
                                                   float* __restrict__ Cf32,
                                                   half_t* __restrict__ Cf16,
                                                   int M, int N, int K) {
    __shared__ __align__(16) half_t sA[2][64 * 40];
    __shared__ __align__(16) half_t sB[2][128 * 40];

    const int tid  = threadIdx.x;
    const int lane = tid & 31, wv = tid >> 5;
    const int wy = wv >> 2, wx = wv & 3;  // 2x4 wave grid
    const int bm = blockIdx.y * 64, bn = blockIdx.x * 128;
    const int l = lane & 15, hi = lane >> 4;

    v8f acc[2][2] = {};

#ifdef HAVE_TDM
    const int wvu = __builtin_amdgcn_readfirstlane(wv);  // scalar branch below
    auto stage = [&](int buf, int k0) {
        if (wvu == 0) {
            tdm_load_2d(&sA[buf][0], &A[(size_t)bm * K + k0], 32, 64,
                        K - k0, M - bm, K);
            tdm_load_2d(&sB[buf][0], &Bt[(size_t)bn * K + k0], 32, 128,
                        K - k0, N - bn, K);
        }
    };
#else
    // staging: A tile 64x32 (1 b128/thread), B tile 128x32 (2/thread)
    const int ar = tid >> 2, ac = (tid & 3) * 8;
    auto stage = [&](int buf, int k0) {
#if USE_ASYNC_LDS
        async_copy_b128(&sA[buf][ar * 40 + ac], &A[(size_t)(bm + ar) * K + k0 + ac]);
#pragma unroll
        for (int j = 0; j < 2; ++j) {
            const int idx = tid + 256 * j;
            const int br = idx >> 2, bc = (idx & 3) * 8;
            async_copy_b128(&sB[buf][br * 40 + bc],
                            &Bt[(size_t)(bn + br) * K + k0 + bc]);
        }
#else
        *(uint4*)&sA[buf][ar * 40 + ac] =
            *(const uint4*)&A[(size_t)(bm + ar) * K + k0 + ac];
#pragma unroll
        for (int j = 0; j < 2; ++j) {
            const int idx = tid + 256 * j;
            const int br = idx >> 2, bc = (idx & 3) * 8;
            *(uint4*)&sB[buf][br * 40 + bc] =
                *(const uint4*)&Bt[(size_t)(bn + br) * K + k0 + bc];
        }
#endif
    };
#endif

    auto compute = [&](int buf) {
        v16h af[2], bf[2];
#pragma unroll
        for (int mt = 0; mt < 2; ++mt) {
            const half_t* base = &sA[buf][(wy * 32 + mt * 16 + l) * 40];
            af[mt] = join16(*(const v8h*)(base + 8 * hi),
                            *(const v8h*)(base + 16 + 8 * hi));
        }
#pragma unroll
        for (int nt = 0; nt < 2; ++nt) {
            const half_t* base = &sB[buf][(wx * 32 + nt * 16 + l) * 40];
            bf[nt] = join16(*(const v8h*)(base + 8 * hi), *(const v8h*)(base + 16 + 8 * hi));
        }
#pragma unroll
        for (int mt = 0; mt < 2; ++mt)
#pragma unroll
            for (int nt = 0; nt < 2; ++nt)
                acc[mt][nt] = __builtin_amdgcn_wmma_f32_16x16x32_f16(
                    false, af[mt], false, bf[nt], (short)0, acc[mt][nt],
                    false, false);
    };

    const int nk = K >> 5;
    stage(0, 0);
    for (int i = 0; i < nk; ++i) {
#ifdef HAVE_TDM
        wait_tensor0();       // no-op for waves with TENSORcnt==0
#elif USE_ASYNC_LDS
        wait_async0();
#endif
        __syncthreads();
        if (i + 1 < nk) stage((i + 1) & 1, (i + 1) << 5);
#if !defined(HAVE_TDM) && !USE_ASYNC_LDS
        __syncthreads();
#endif
        compute(i & 1);
#if !defined(HAVE_TDM) && !USE_ASYNC_LDS
        __syncthreads();
#endif
    }

    // epilogue (branchless; compile-time variants)
#pragma unroll
    for (int mt = 0; mt < 2; ++mt) {
#pragma unroll
        for (int nt = 0; nt < 2; ++nt) {
            const int col = bn + wx * 32 + nt * 16 + l;
            const float bv = bias[col];
            const int row0 = bm + wy * 32 + mt * 16 + 8 * hi;
            size_t base = (size_t)row0 * N + col;
#pragma unroll
            for (int r = 0; r < 8; ++r) {
                float v = acc[mt][nt][r] + bv;
                if constexpr (RES) v += residual[base];
                if constexpr (OUT32) Cf32[base] = v;
                if constexpr (OUT16) Cf16[base] = (half_t)v;
                base += (size_t)N;
            }
        }
    }
}

// ---------------------------------------------------------------------------
// RoPE + repack: qkv fp32 [B*T,3*DIM] -> Q[B,H,T,64], K[B,H,T,64], V^T[B,H,64,T]
// ---------------------------------------------------------------------------
__global__ void rope_pack_kernel(const float* __restrict__ qkv,
                                 const float* __restrict__ rsin,
                                 const float* __restrict__ rcos,
                                 half_t* __restrict__ Qo,
                                 half_t* __restrict__ Ko,
                                 half_t* __restrict__ VTo) {
    const size_t i = (size_t)blockIdx.x * blockDim.x + threadIdx.x;  // B*H*T*64
    const int d = (int)(i & 63);
    const int t = (int)((i >> 6) & (TT - 1));
    const int h = (int)(((i >> 6) / TT) & (NHEAD - 1));
    const int b = (int)((i >> 6) / ((size_t)TT * NHEAD));
    const float* row = qkv + (size_t)(b * TT + t) * (3 * DIM);
    const int c = h * HDIM + d;
    const float qv = row[c], kv = row[DIM + c], vv = row[2 * DIM + c];
    const int dp = (d < 32) ? (d + 32) : (d - 32);
    const float sgn = (d < 32) ? -1.f : 1.f;
    const float qr = row[h * HDIM + dp] * sgn;
    const float kr = row[DIM + h * HDIM + dp] * sgn;
    const float sn = rsin[t * HDIM + d], cs = rcos[t * HDIM + d];
    const size_t bh = (size_t)(b * NHEAD + h);
    Qo[(bh * TT + t) * HDIM + d]  = (half_t)(qv * cs + qr * sn);
    Ko[(bh * TT + t) * HDIM + d]  = (half_t)(kv * cs + kr * sn);
    VTo[(bh * HDIM + d) * TT + t] = (half_t)vv;
}

// ---------------------------------------------------------------------------
// Flash attention: one wave per 16-row Q tile; 32 keys per iteration.
// ---------------------------------------------------------------------------
__global__ __launch_bounds__(128) void attn_kernel(const half_t* __restrict__ Q,
                                                   const half_t* __restrict__ Kk,
                                                   const half_t* __restrict__ VT,
                                                   half_t* __restrict__ Y) {
    __shared__ __align__(16) half_t sP[4][16 * 32];
    const int lane = threadIdx.x & 31, wv = threadIdx.x >> 5;
    const int l = lane & 15, hi = lane >> 4;
    const int h = blockIdx.y, b = blockIdx.z;
    const int qi = blockIdx.x * 4 + wv;
    const int tq = qi * 16;
    const size_t bh = (size_t)(b * NHEAD + h);
    const half_t* qbase  = Q  + bh * TT * HDIM;
    const half_t* kbase  = Kk + bh * TT * HDIM;
    const half_t* vtbase = VT + bh * HDIM * TT;
    half_t* myP = &sP[wv][0];

    v16h aq[2];
    {
        const half_t* qr = qbase + (size_t)(tq + l) * HDIM;
#pragma unroll
        for (int ch = 0; ch < 2; ++ch)
            aq[ch] = join16(*(const v8h*)(qr + ch * 32 + 8 * hi),
                            *(const v8h*)(qr + ch * 32 + 16 + 8 * hi));
    }

    v8f O[4] = {};
    float mstat[8], lstat[8];
#pragma unroll
    for (int r = 0; r < 8; ++r) { mstat[r] = -1e30f; lstat[r] = 0.f; }

    const float scale = 0.125f;     // 1/sqrt(64)
    const int nkb = (qi + 2) >> 1;  // 32-key blocks needed (causal)
    for (int j = 0; j < nkb; ++j) {
        const int kk0 = j * 32;
        v8f S[2] = {};
#pragma unroll
        for (int nt = 0; nt < 2; ++nt) {
            const half_t* kr = kbase + (size_t)(kk0 + nt * 16 + l) * HDIM;
#pragma unroll
            for (int ch = 0; ch < 2; ++ch) {
                v16h bk = join16(*(const v8h*)(kr + ch * 32 + 8 * hi),
                                 *(const v8h*)(kr + ch * 32 + 16 + 8 * hi));
                S[nt] = __builtin_amdgcn_wmma_f32_16x16x32_f16(
                    false, aq[ch], false, bk, (short)0, S[nt], false, false);
            }
        }
#pragma unroll
        for (int r = 0; r < 8; ++r) {
            const int qrow = tq + r + 8 * hi;
            float s0 = S[0][r] * scale;
            float s1 = S[1][r] * scale;
            if (kk0 + l > qrow)      s0 = -1e30f;
            if (kk0 + 16 + l > qrow) s1 = -1e30f;
            float rm = fmaxf(s0, s1);
            for (int m = 8; m >= 1; m >>= 1) rm = fmaxf(rm, __shfl_xor(rm, m, 32));
            const float mnew  = fmaxf(mstat[r], rm);
            const float alpha = __expf(mstat[r] - mnew);
            const float p0 = __expf(s0 - mnew);
            const float p1 = __expf(s1 - mnew);
            float rs = p0 + p1;
            for (int m = 8; m >= 1; m >>= 1) rs += __shfl_xor(rs, m, 32);
            lstat[r] = lstat[r] * alpha + rs;
            mstat[r] = mnew;
#pragma unroll
            for (int nt = 0; nt < 4; ++nt) O[nt][r] *= alpha;
            myP[(r + 8 * hi) * 32 + l]      = (half_t)p0;
            myP[(r + 8 * hi) * 32 + 16 + l] = (half_t)p1;
        }
        const half_t* pr = myP + l * 32;
        v16h ap = join16(*(const v8h*)(pr + 8 * hi),
                         *(const v8h*)(pr + 16 + 8 * hi));
#pragma unroll
        for (int nt = 0; nt < 4; ++nt) {
            const half_t* vr = vtbase + (size_t)(nt * 16 + l) * TT + kk0;
            v16h bv = join16(*(const v8h*)(vr + 8 * hi), *(const v8h*)(vr + 16 + 8 * hi));
            O[nt] = __builtin_amdgcn_wmma_f32_16x16x32_f16(
                false, ap, false, bv, (short)0, O[nt], false, false);
        }
    }
#pragma unroll
    for (int nt = 0; nt < 4; ++nt) {
#pragma unroll
        for (int r = 0; r < 8; ++r) {
            const int trow = tq + r + 8 * hi;
            const float val = O[nt][r] / lstat[r];
            Y[(size_t)(b * TT + trow) * DIM + h * HDIM + nt * 16 + l] = (half_t)val;
        }
    }
}

// ---------------------------------------------------------------------------
// SwiGLU gate: a = silu(g1) * g2, all f16
// ---------------------------------------------------------------------------
__global__ void silu_mul_kernel(const half_t* __restrict__ g1,
                                const half_t* __restrict__ g2,
                                half_t* __restrict__ a, int n) {
    int i = blockIdx.x * blockDim.x + threadIdx.x;
    if (i < n) {
        float x = (float)g1[i];
        float s = x / (1.f + __expf(-x));
        a[i] = (half_t)(s * (float)g2[i]);
    }
}

// ---------------------------------------------------------------------------
extern "C" void kernel_launch(void* const* d_in, const int* in_sizes, int n_in,
                              void* d_out, int out_size, void* d_ws, size_t ws_size,
                              hipStream_t stream) {
    const float* x     = (const float*)d_in[0];
    // d_in[1] = mask (causal, handled analytically)
    const float* rsin  = (const float*)d_in[2];
    const float* rcos  = (const float*)d_in[3];
    const float* n1w   = (const float*)d_in[4];
    const float* n2w   = (const float*)d_in[5];
    const float* qkvw  = (const float*)d_in[6];
    const float* qkvb  = (const float*)d_in[7];
    const float* projw = (const float*)d_in[8];
    const float* projb = (const float*)d_in[9];
    const float* w1    = (const float*)d_in[10];
    const float* b1    = (const float*)d_in[11];
    const float* w2    = (const float*)d_in[12];
    const float* b2    = (const float*)d_in[13];
    const float* w3    = (const float*)d_in[14];
    const float* b3    = (const float*)d_in[15];
    float* out = (float*)d_out;

    char* p = (char*)d_ws;
    auto alloc = [&](size_t bytes) -> char* {
        char* r = p;
        p += (bytes + 255) & ~(size_t)255;
        return r;
    };
    half_t* qkvw_t = (half_t*)alloc((size_t)DIM * 3 * DIM * 2);  // [3*DIM, DIM]
    half_t* projw_t = (half_t*)alloc((size_t)DIM * DIM * 2);     // [DIM, DIM]
    half_t* w1_t = (half_t*)alloc((size_t)FFN * DIM * 2);        // [FFN, DIM]
    half_t* w2_t = (half_t*)alloc((size_t)FFN * DIM * 2);        // [FFN, DIM]
    half_t* w3_t = (half_t*)alloc((size_t)DIM * FFN * 2);        // [DIM, FFN]
    half_t* h1 = (half_t*)alloc((size_t)MROWS * DIM * 2);
    float*  qkv = (float*)alloc((size_t)MROWS * 3 * DIM * 4);
    half_t* qh = (half_t*)alloc((size_t)BB * NHEAD * TT * HDIM * 2);
    half_t* kh = (half_t*)alloc((size_t)BB * NHEAD * TT * HDIM * 2);
    half_t* vth = (half_t*)alloc((size_t)BB * NHEAD * HDIM * TT * 2);
    half_t* yh = (half_t*)alloc((size_t)MROWS * DIM * 2);
    float*  x2 = (float*)alloc((size_t)MROWS * DIM * 4);
    half_t* h2 = (half_t*)alloc((size_t)MROWS * DIM * 2);
    half_t* g1h = (half_t*)alloc((size_t)MROWS * FFN * 2);
    half_t* g2h = (half_t*)alloc((size_t)MROWS * FFN * 2);
    half_t* ah  = (half_t*)alloc((size_t)MROWS * FFN * 2);

    auto ctr = [&](const float* w, half_t* wt, int K, int N) {
        cast_transpose_kernel<<<dim3(K / 32, N / 32), 256, 0, stream>>>(w, wt, K, N);
    };
    ctr(qkvw, qkvw_t, DIM, 3 * DIM);
    ctr(projw, projw_t, DIM, DIM);
    ctr(w1, w1_t, DIM, FFN);
    ctr(w2, w2_t, DIM, FFN);
    ctr(w3, w3_t, FFN, DIM);

    rmsnorm_kernel<<<MROWS, 256, 0, stream>>>(x, n1w, h1);

    gemm_kernel<false, true, false><<<dim3(3 * DIM / 128, MROWS / 64), 256, 0, stream>>>(
        h1, qkvw_t, qkvb, nullptr, qkv, nullptr, MROWS, 3 * DIM, DIM);

    rope_pack_kernel<<<(BB * NHEAD * TT * HDIM) / 256, 256, 0, stream>>>(
        qkv, rsin, rcos, qh, kh, vth);

    attn_kernel<<<dim3(TT / 64, NHEAD, BB), 128, 0, stream>>>(qh, kh, vth, yh);

    gemm_kernel<true, true, false><<<dim3(DIM / 128, MROWS / 64), 256, 0, stream>>>(
        yh, projw_t, projb, x, x2, nullptr, MROWS, DIM, DIM);

    rmsnorm_kernel<<<MROWS, 256, 0, stream>>>(x2, n2w, h2);

    gemm_kernel<false, false, true><<<dim3(FFN / 128, MROWS / 64), 256, 0, stream>>>(
        h2, w1_t, b1, nullptr, nullptr, g1h, MROWS, FFN, DIM);
    gemm_kernel<false, false, true><<<dim3(FFN / 128, MROWS / 64), 256, 0, stream>>>(
        h2, w2_t, b2, nullptr, nullptr, g2h, MROWS, FFN, DIM);

    silu_mul_kernel<<<(MROWS * FFN) / 256, 256, 0, stream>>>(g1h, g2h, ah,
                                                             MROWS * FFN);

    gemm_kernel<true, true, false><<<dim3(DIM / 128, MROWS / 64), 256, 0, stream>>>(
        ah, w3_t, b3, x2, out, nullptr, MROWS, DIM, FFN);
}
